// Model_71717363909022
// MI455X (gfx1250) — matrix operand order repaired and checked by reading, not compiled, not run on hardware
//
#include <hip/hip_runtime.h>
#include <hip/hip_bf16.h>

// Problem constants (from reference): SEQ=512, BATCH=256, IN=784, HID=10
#define R_SEQ   512
#define R_BATCH 256
#define R_IN    784
#define R_HID   10

typedef __attribute__((ext_vector_type(16))) __bf16 v16bf;
typedef __attribute__((ext_vector_type(8)))  float  v8f;

// Hardware packed fp32 -> bf16 (RNE): one VOP3 per dword instead of ~10 VALU
// ops of manual bit twiddling.
__device__ __forceinline__ unsigned int cvt_pk_bf16(float lo, float hi) {
    unsigned int d;
    asm("v_cvt_pk_bf16_f32 %0, %1, %2" : "=v"(d) : "v"(lo), "v"(hi));
    return d;
}

// scalar helper for the LDS fill (rate-irrelevant: runs once per block)
__device__ __forceinline__ unsigned short f2bf(float f) {
    unsigned int u = __float_as_uint(f);
    return (unsigned short)((u + 0x7FFFu + ((u >> 16) & 1u)) >> 16);
}

// ---------------------------------------------------------------------------
// Phase 1: xi[m, n] = x[m, :] . W_ih[n, :] + (b_ih[n] + b_hh[n])
//   m in [0, SEQ*BATCH), n in [0, HID).  Written directly into d_out's
//   `out` region (unpadded [m][10]); phase 2 overwrites it in place.
// One wave per 16(M)x16(N) tile; K consumed 32 at a time via
// v_wmma_f32_16x16x32_bf16.  W_ih staged once per block into LDS as bf16,
// padded to [16][800] so the K tail (784..799) and rows n=10..15 are zeros.
// Main K loop (24 iters) is branch-free; the half-chunk tail is peeled.
// ---------------------------------------------------------------------------
__global__ __launch_bounds__(256) void rnn_xi_gemm(
    const float* __restrict__ x,
    const float* __restrict__ W_ih,
    const float* __restrict__ b_ih,
    const float* __restrict__ b_hh,
    float* __restrict__ out)
{
    __shared__ unsigned short ldsB[16 * 800];   // 25.6 KB bf16, zero-padded

    const int tid = threadIdx.x;
    for (int idx = tid; idx < 16 * 800; idx += 256) {
        const int n = idx / 800;
        const int k = idx - n * 800;
        float v = (n < R_HID && k < R_IN) ? W_ih[n * R_IN + k] : 0.0f;
        ldsB[idx] = f2bf(v);
    }
    __syncthreads();

    const int lane = tid & 31;
    const int wave = tid >> 5;
    const int tile = blockIdx.x * 8 + wave;          // 8192 tiles total
    const int g    = lane >> 4;                      // lane half-group
    const int nl   = lane & 15;                      // N column / A row sel

    // A row for this lane; K groups: g*8 + {0..7} and 16 + g*8 + {0..7}
    const float4* __restrict__ arow =
        (const float4*)(x + (long)(tile * 16 + nl) * R_IN + g * 8);
    // B row in LDS: 8 bf16 at kt*32 + g*8 and 8 at kt*32 + 16 + g*8 (16B aligned)
    const uint4* __restrict__ brow =
        (const uint4*)&ldsB[nl * 800 + g * 8];

    union { v16bf v; unsigned int d[8]; } afrag, bfrag;
    v8f acc = {};

    #pragma unroll 4
    for (int kt = 0; kt < 24; ++kt) {                // branch-free main body
        const float4* ap = arow + kt * 8;            // kt*32 floats
        float4 a0 = ap[0];
        float4 a1 = ap[1];
        float4 a2 = ap[4];                           // +16 floats
        float4 a3 = ap[5];
        afrag.d[0] = cvt_pk_bf16(a0.x, a0.y); afrag.d[1] = cvt_pk_bf16(a0.z, a0.w);
        afrag.d[2] = cvt_pk_bf16(a1.x, a1.y); afrag.d[3] = cvt_pk_bf16(a1.z, a1.w);
        afrag.d[4] = cvt_pk_bf16(a2.x, a2.y); afrag.d[5] = cvt_pk_bf16(a2.z, a2.w);
        afrag.d[6] = cvt_pk_bf16(a3.x, a3.y); afrag.d[7] = cvt_pk_bf16(a3.z, a3.w);

        const uint4* bp = brow + kt * 4;             // kt*32 bf16 = kt*4 uint4
        uint4 b0 = bp[0], b1 = bp[2];
        bfrag.d[0] = b0.x; bfrag.d[1] = b0.y; bfrag.d[2] = b0.z; bfrag.d[3] = b0.w;
        bfrag.d[4] = b1.x; bfrag.d[5] = b1.y; bfrag.d[6] = b1.z; bfrag.d[7] = b1.w;

        acc = __builtin_amdgcn_wmma_f32_16x16x32_bf16(
            false, afrag.v, false, bfrag.v, (short)0, acc, false, false);
    }

    {   // ---- peeled tail: kt=24, K = 768..783 valid; 784..799 are zero
        const float4* ap = arow + 24 * 8;
        float4 a0 = ap[0];
        float4 a1 = ap[1];
        afrag.d[0] = cvt_pk_bf16(a0.x, a0.y); afrag.d[1] = cvt_pk_bf16(a0.z, a0.w);
        afrag.d[2] = cvt_pk_bf16(a1.x, a1.y); afrag.d[3] = cvt_pk_bf16(a1.z, a1.w);
        afrag.d[4] = 0u; afrag.d[5] = 0u; afrag.d[6] = 0u; afrag.d[7] = 0u;

        const uint4* bp = brow + 24 * 4;             // LDS padded with zeros
        uint4 b0 = bp[0], b1 = bp[2];
        bfrag.d[0] = b0.x; bfrag.d[1] = b0.y; bfrag.d[2] = b0.z; bfrag.d[3] = b0.w;
        bfrag.d[4] = b1.x; bfrag.d[5] = b1.y; bfrag.d[6] = b1.z; bfrag.d[7] = b1.w;

        acc = __builtin_amdgcn_wmma_f32_16x16x32_bf16(
            false, afrag.v, false, bfrag.v, (short)0, acc, false, false);
    }

    // ---- Epilogue: bias + store valid columns (n < 10), unpadded rows of 10
    if (nl < R_HID) {
        const float bias = b_ih[nl] + b_hh[nl];
        #pragma unroll
        for (int r = 0; r < 8; ++r) {
            const long m = (long)tile * 16 + r + g * 8;   // C layout: M=r+g*8
            out[m * R_HID + nl] = acc[r] + bias;
        }
    }
}

// ---------------------------------------------------------------------------
// Phase 2: sequential scan over S; one lane per batch chain (256 lanes total,
// 8 wave32 blocks spread over 8 WGPs).  W_hh held in registers.  Reads xi row
// from d_out and overwrites it in place with h_t; h_n appended at the end.
// ---------------------------------------------------------------------------
__global__ __launch_bounds__(32) void rnn_scan(
    const float* __restrict__ W_hh,
    float* __restrict__ out)
{
    const int b = blockIdx.x * 32 + threadIdx.x;     // batch index 0..255

    float w[R_HID * R_HID];
    #pragma unroll
    for (int i = 0; i < 25; ++i) {                   // 100 floats = 25 float4
        float4 t = *(const float4*)(W_hh + i * 4);
        w[i * 4 + 0] = t.x; w[i * 4 + 1] = t.y;
        w[i * 4 + 2] = t.z; w[i * 4 + 3] = t.w;
    }

    float h[R_HID];
    #pragma unroll
    for (int j = 0; j < R_HID; ++j) h[j] = 0.0f;

    for (int s = 0; s < R_SEQ; ++s) {
        float* __restrict__ row = out + ((long)s * R_BATCH + b) * R_HID;
        float xi[R_HID];
        #pragma unroll
        for (int j = 0; j < 5; ++j) {                // rows are 8B aligned
            float2 t = *(const float2*)(row + j * 2);
            xi[j * 2] = t.x; xi[j * 2 + 1] = t.y;
        }
        float hn[R_HID];
        #pragma unroll
        for (int j = 0; j < R_HID; ++j) {
            float a = xi[j];
            #pragma unroll
            for (int k = 0; k < R_HID; ++k)
                a = fmaf(w[j * R_HID + k], h[k], a);
            hn[j] = fmaxf(a, 0.0f);
        }
        #pragma unroll
        for (int j = 0; j < R_HID; ++j) h[j] = hn[j];
        #pragma unroll
        for (int j = 0; j < 5; ++j) {
            float2 t; t.x = h[j * 2]; t.y = h[j * 2 + 1];
            *(float2*)(row + j * 2) = t;
        }
    }

    float* __restrict__ hn_out =
        out + (long)R_SEQ * R_BATCH * R_HID + (long)b * R_HID;
    #pragma unroll
    for (int j = 0; j < 5; ++j) {
        float2 t; t.x = h[j * 2]; t.y = h[j * 2 + 1];
        *(float2*)(hn_out + j * 2) = t;
    }
}

extern "C" void kernel_launch(void* const* d_in, const int* in_sizes, int n_in,
                              void* d_out, int out_size, void* d_ws, size_t ws_size,
                              hipStream_t stream) {
    (void)in_sizes; (void)n_in; (void)out_size; (void)d_ws; (void)ws_size;
    const float* x    = (const float*)d_in[0];
    const float* W_ih = (const float*)d_in[1];
    const float* W_hh = (const float*)d_in[2];
    const float* b_ih = (const float*)d_in[3];
    const float* b_hh = (const float*)d_in[4];
    float* out = (float*)d_out;

    // SEQ*BATCH/16 = 8192 tiles, 8 waves/block -> 1024 blocks
    rnn_xi_gemm<<<1024, 256, 0, stream>>>(x, W_ih, b_ih, b_hh, out);
    // 256 batch chains, 1 lane each -> 8 single-wave blocks
    rnn_scan<<<8, 32, 0, stream>>>(W_hh, out);
}
